// SurrogateSNN_20186346291328
// MI455X (gfx1250) — compile-verified, matrix operand order checked by reading, not run on hardware
//
#include <hip/hip_runtime.h>
#include <hip/hip_bf16.h>

typedef __attribute__((ext_vector_type(16))) _Float16     v16h;
typedef __attribute__((ext_vector_type(8)))  _Float16     v8h;
typedef __attribute__((ext_vector_type(8)))  float        v8f;
typedef __attribute__((ext_vector_type(4)))  unsigned int u32x4;
typedef __attribute__((ext_vector_type(4)))  int          i32x4;
typedef __attribute__((ext_vector_type(8)))  int          i32x8;

#define BETA 0.9f
#define THR  1.0f
#define NB   128   // batch

#if defined(__has_builtin)
#if __has_builtin(__builtin_amdgcn_tensor_load_to_lds) && __has_builtin(__builtin_amdgcn_s_wait_tensorcnt)
#define USE_TDM 1
#endif
#endif
#ifndef USE_TDM
#define USE_TDM 0
#endif

// ---------------------------------------------------------------- utilities

__global__ void k_zero(float* __restrict__ p, long n) {
  long i = (long)blockIdx.x * blockDim.x + threadIdx.x;
  if (i < n) p[i] = 0.f;
}

// src is [Nreal, Kreal] f32 row-major -> dst [Npad, Kpad] f16, zero padded
__global__ void k_cvt_pad(const float* __restrict__ src, _Float16* __restrict__ dst,
                          int Nreal, int Npad, int Kreal, int Kpad) {
  int i = blockIdx.x * blockDim.x + threadIdx.x;
  if (i >= Npad * Kpad) return;
  int k = i % Kpad, r = i / Kpad;
  float v = (r < Nreal && k < Kreal) ? src[r * Kreal + k] : 0.f;
  dst[i] = (_Float16)v;
}

// rate-coded Bernoulli spike generation, deterministic hash RNG per (t, idx)
__global__ void k_spikegen(const float* __restrict__ x, _Float16* __restrict__ out,
                           int t, int n) {
  int i = blockIdx.x * blockDim.x + threadIdx.x;
  if (i >= n) return;
  float p = (x[i] + 1.f) * 0.5f;
  p = fminf(fmaxf(p, 0.f), 1.f);
  unsigned h = (unsigned)i * 0x9E3779B9u ^ ((unsigned)t * 0x85EBCA6Bu + 0xC2B2AE35u);
  h ^= h >> 16; h *= 0x7FEB352Du; h ^= h >> 15; h *= 0x846CA68Bu; h ^= h >> 16;
  float u = (float)(h >> 8) * (1.0f / 16777216.0f);
  out[i] = (_Float16)((u < p) ? 1.f : 0.f);
}

// ------------------------------------------------ implicit-im2col conv GEMM
// One wave32 computes a 16(M=pixels) x 16(N=out-chan) tile via WMMA f16.
template<int CIN, int HH, int WW, int KPAD>
__global__ void k_conv_wmma(const _Float16* __restrict__ inp,
                            const _Float16* __restrict__ wgt,
                            const float*    __restrict__ bias,
                            float*          __restrict__ outp,
                            int cout) {
  const int lane  = threadIdx.x & 31;
  const int wid   = threadIdx.x >> 5;
  const int tilesN = cout >> 4;
  const int tile  = blockIdx.x * (blockDim.x >> 5) + wid;
  const int tm = tile / tilesN, tn = tile % tilesN;

  const int ra  = tm * 16 + (lane & 15);            // im2col row = output pixel
  const int xx  = ra % WW;
  const int tmp = ra / WW;
  const int yy  = tmp % HH;
  const int bb  = tmp / HH;
  const _Float16* ibase = inp + (long)bb * CIN * HH * WW;

  const int grp = (lane >> 4) << 3;                 // K sub-group offset: 0 or 8
  const int nr  = tn * 16 + (lane & 15);
  const _Float16* wrow = wgt + (long)nr * KPAD;

  constexpr int KREAL = CIN * 9;
  v8f acc = {0.f, 0.f, 0.f, 0.f, 0.f, 0.f, 0.f, 0.f};

  for (int k0 = 0; k0 < KPAD; k0 += 32) {
    v16h a, w;
#pragma unroll
    for (int j = 0; j < 16; ++j) {
      const int k = k0 + grp + (j & 7) + ((j >> 3) << 4);   // {0..7,16..23}+grp
      _Float16 av = (_Float16)0.f;
      if (k < KREAL) {
        const int ci = k / 9;
        const int r  = k - ci * 9;
        const int ky = r / 3;
        const int kx = r - ky * 3;
        const int iy = yy + ky - 1, ix = xx + kx - 1;
        if ((unsigned)iy < (unsigned)HH && (unsigned)ix < (unsigned)WW)
          av = ibase[(ci * HH + iy) * WW + ix];
      }
      a[j] = av;
      w[j] = wrow[k];                                // zero-padded beyond KREAL
    }
    acc = __builtin_amdgcn_wmma_f32_16x16x32_f16(false, a, false, w,
                                                 (short)0, acc, false, false);
  }

  const int colc = tn * 16 + (lane & 15);
  const float bv = bias[colc];
#pragma unroll
  for (int j = 0; j < 8; ++j) {
    const int row = tm * 16 + ((lane >> 4) << 3) + j;
    const int x2 = row % WW;
    const int t2 = row / WW;
    const int y2 = t2 % HH;
    const int b2 = t2 / HH;
    outp[(((long)b2 * cout + colc) * HH + y2) * WW + x2] = acc[j] + bv;
  }
}

// -------------------------------------------------------------- FC WMMA GEMM
// One block owns M-tile tm = blockIdx.x (16 rows). The 16 x K activation slab
// is staged into LDS once per K-chunk via the Tensor Data Mover (TENSORcnt),
// with TDM pad (+4 DWORDs / row) so the 16 rows cover all 64 LDS banks.
// 4 waves then sweep the N-tiles: wave w handles tn = w, w+4, ...
__global__ void k_fc_wmma(const _Float16* __restrict__ A,   // [M][K]
                          const _Float16* __restrict__ Wt,  // [Npad][K]
                          const float*    __restrict__ bias,
                          float*          __restrict__ outp,
                          int Npad, int K, int KC, int padLog,
                          int ldc, int ncols) {
  __shared__ _Float16 lds[16 * 520];                // max: KC=512 -> RS=520
  const int lane = threadIdx.x & 31;
  const int wid  = threadIdx.x >> 5;
  const int tilesN = Npad >> 4;
  const int tm  = blockIdx.x;
  const int RS  = KC + 8;                           // padded LDS row stride (halves)
  const int row = lane & 15;
  const int grp = (lane >> 4) << 3;

  v8f accv[4];
  const v8f vz = {0.f, 0.f, 0.f, 0.f, 0.f, 0.f, 0.f, 0.f};
  accv[0] = vz; accv[1] = vz; accv[2] = vz; accv[3] = vz;

  for (int kc = 0; kc < K; kc += KC) {
#if USE_TDM
    if (threadIdx.x < 32) {                         // wave 0 drives the TDM
      unsigned ldsaddr = (unsigned)(size_t)(&lds[0]);
      unsigned long long ga =
          (unsigned long long)(size_t)(A + (size_t)tm * 16 * K + kc);
      u32x4 g0 = { 1u,                              // count=1 (valid), user mode
                   ldsaddr,                         // LDS dest byte address
                   (unsigned)(ga & 0xFFFFFFFFu),
                   (unsigned)((ga >> 32) & 0x01FFFFFFu) | 0x80000000u }; // type=2
      const unsigned td0 = (unsigned)K;             // tensor dim0 (elements)
      const unsigned td1 = 0x100000u;               // generous dim1 bound
      // data_size=1(2B) @16, pad_enable @20, pad_interval @22, pad_amount=3 @25
      const unsigned cfg = (1u << 16) | (1u << 20) |
                           ((unsigned)padLog << 22) | (3u << 25);
      i32x8 g1 = { (int)cfg,
                   (int)((td0 & 0xFFFFu) << 16),                 // td0[15:0]
                   (int)((td0 >> 16) | ((td1 & 0xFFFFu) << 16)), // td0[31:16],td1 lo
                   (int)((td1 >> 16) | ((unsigned)KC << 16)),    // td1 hi, tile_dim0
                   (int)16,                                      // tile_dim1=16
                   (int)td0,                                     // dim0 stride = K
                   0, 0 };
      i32x4 gz = {0, 0, 0, 0};
#if __clang_major__ >= 23
      i32x8 gz8 = {0, 0, 0, 0, 0, 0, 0, 0};
      __builtin_amdgcn_tensor_load_to_lds(g0, g1, gz, gz, gz8, 0);
#else
      __builtin_amdgcn_tensor_load_to_lds(g0, g1, gz, gz, 0);
#endif
      __builtin_amdgcn_s_wait_tensorcnt(0);
    }
#else
    for (int idx = threadIdx.x; idx < 16 * KC; idx += blockDim.x) {
      int r = idx / KC, c = idx - r * KC;
      lds[r * RS + c] = A[(size_t)(tm * 16 + r) * K + kc + c];
    }
#endif
    __syncthreads();

    int i = 0;
    for (int tn = wid; tn < tilesN; tn += 4, ++i) {
      const _Float16* wrow = Wt + (size_t)(tn * 16 + row) * K + kc;
      v8f acc = accv[i];
      for (int k0 = 0; k0 < KC; k0 += 32) {
        if (k0 + 32 < KC) __builtin_prefetch(wrow + k0 + 32, 0, 3);
        v8h alo = *(const v8h*)&lds[row * RS + k0 + grp];
        v8h ahi = *(const v8h*)&lds[row * RS + k0 + 16 + grp];
        v8h wlo = *(const v8h*)(wrow + k0 + grp);
        v8h whi = *(const v8h*)(wrow + k0 + 16 + grp);
        v16h a, w;
#pragma unroll
        for (int j = 0; j < 8; ++j) {
          a[j] = alo[j]; a[8 + j] = ahi[j];
          w[j] = wlo[j]; w[8 + j] = whi[j];
        }
        acc = __builtin_amdgcn_wmma_f32_16x16x32_f16(false, a, false, w,
                                                     (short)0, acc, false, false);
      }
      accv[i] = acc;
    }
    __syncthreads();
  }

  int i = 0;
  for (int tn = wid; tn < tilesN; tn += 4, ++i) {
    const int col = tn * 16 + (lane & 15);
    const float bv = (col < ncols) ? bias[col] : 0.f;
#pragma unroll
    for (int j = 0; j < 8; ++j) {
      const int orow = tm * 16 + ((lane >> 4) << 3) + j;
      if (col < ncols) outp[(long)orow * ldc + col] = accv[i][j] + bv;
    }
  }
}

// -------------------------------------------------- LIF + reductions + pool

__device__ __forceinline__ void block_reduce_add(float s, float* accum) {
  __shared__ float red[256];
  red[threadIdx.x] = s;
  __syncthreads();
  for (int st = blockDim.x >> 1; st > 0; st >>= 1) {
    if ((int)threadIdx.x < st) red[threadIdx.x] += red[threadIdx.x + st];
    __syncthreads();
  }
  if (threadIdx.x == 0) atomicAdd(accum, red[0]);
}

__global__ void k_lif(const float* __restrict__ cur, float* __restrict__ mem,
                      _Float16* __restrict__ spk, float* __restrict__ spikeAcc,
                      int n) {
  int i = blockIdx.x * blockDim.x + threadIdx.x;
  float s = 0.f;
  if (i < n) {
    float m = mem[i];
    float reset = (m - THR > 0.f) ? 1.f : 0.f;       // detached subtract-reset
    m = BETA * m + cur[i] - reset * THR;
    mem[i] = m;
    s = (m - THR > 0.f) ? 1.f : 0.f;
    spk[i] = (_Float16)s;
  }
  block_reduce_add(s, spikeAcc);
}

// final layer: also record f32 spikes + membranes into d_out
__global__ void k_lif_out(const float* __restrict__ cur, float* __restrict__ mem,
                          float* __restrict__ spk_rec_t, float* __restrict__ mem_rec_t,
                          float* __restrict__ spikeAcc, int n) {
  int i = blockIdx.x * blockDim.x + threadIdx.x;
  float s = 0.f;
  if (i < n) {
    float m = mem[i];
    float reset = (m - THR > 0.f) ? 1.f : 0.f;
    m = BETA * m + cur[i] - reset * THR;
    mem[i] = m;
    s = (m - THR > 0.f) ? 1.f : 0.f;
    spk_rec_t[i] = s;
    mem_rec_t[i] = m;
  }
  block_reduce_add(s, spikeAcc);
}

__global__ void k_pool(const _Float16* __restrict__ in, _Float16* __restrict__ out,
                       float* __restrict__ sumAcc, int C, int HO, int WO, int n) {
  int i = blockIdx.x * blockDim.x + threadIdx.x;
  float s = 0.f;
  if (i < n) {
    int xo = i % WO; int t = i / WO;
    int yo = t % HO; t /= HO;
    int c  = t % C;  int b = t / C;
    const int WI = WO * 2;
    const _Float16* p = in + (((long)b * C + c) * (HO * 2) + yo * 2) * WI + xo * 2;
    float m = fmaxf(fmaxf((float)p[0], (float)p[1]),
                    fmaxf((float)p[WI], (float)p[WI + 1]));
    out[i] = (_Float16)m;
    s = m;
  }
  block_reduce_add(s, sumAcc);
}

__global__ void k_finalize(const float* __restrict__ acc, float* __restrict__ tail) {
  if (threadIdx.x == 0) {
    tail[0] = acc[0]; tail[1] = acc[1]; tail[2] = acc[2];
    tail[3] = acc[3]; tail[4] = acc[4]; tail[5] = acc[5];
    tail[6]  = 25.f * 4194304.f;  // B*32*32*32
    tail[7]  = 25.f * 2097152.f;  // B*64*16*16
    tail[8]  = 25.f * 1048576.f;  // B*128*8*8
    tail[9]  = 25.f * 32768.f;    // B*256
    tail[10] = 25.f * 16384.f;    // B*128
    tail[11] = 25.f * 1280.f;     // B*10
    tail[12] = acc[6] * (64.f * 9.f);    // x1.sum * 576
    tail[13] = acc[7] * (128.f * 9.f);   // x2.sum * 1152
    tail[14] = acc[8] * 256.f;           // x3.sum * 256
    tail[15] = acc[3] * 128.f;           // spk4.sum * 128
    tail[16] = acc[4] * 10.f;            // spk5.sum * 10
    tail[17] = 0.f;
  }
}

// ---------------------------------------------------------------- launcher

extern "C" void kernel_launch(void* const* d_in, const int* in_sizes, int n_in,
                              void* d_out, int out_size, void* d_ws, size_t ws_size,
                              hipStream_t stream) {
  (void)in_sizes; (void)n_in; (void)out_size; (void)ws_size;
  const float* x   = (const float*)d_in[0];
  const float* w1  = (const float*)d_in[1];  const float* b1  = (const float*)d_in[2];
  const float* w2  = (const float*)d_in[3];  const float* b2  = (const float*)d_in[4];
  const float* w3  = (const float*)d_in[5];  const float* b3  = (const float*)d_in[6];
  const float* wf1 = (const float*)d_in[7];  const float* bf1 = (const float*)d_in[8];
  const float* wf2 = (const float*)d_in[9];  const float* bf2 = (const float*)d_in[10];
  const float* wf3 = (const float*)d_in[11]; const float* bf3 = (const float*)d_in[12];
  float* out = (float*)d_out;

  char* base = (char*)d_ws;
  size_t off = 0;
  auto F = [&](size_t n) { float* p = (float*)(base + off); off += n * sizeof(float); return p; };
  auto Hh = [&](size_t n) { _Float16* p = (_Float16*)(base + off); off += n * sizeof(_Float16); return p; };

  // membrane state (zeroed each launch, contiguous) + accumulators
  float* mem1 = F(4194304);
  float* mem2 = F(2097152);
  float* mem3 = F(1048576);
  float* mem4 = F(32768);
  float* mem5 = F(16384);
  float* mem6 = F(1280);
  float* acc  = F(16);
  float* cur  = F(4194304);          // reused current buffer (max layer size)
  _Float16* sin_t = Hh(393216);      // spike input for one timestep
  _Float16* spk1  = Hh(4194304);
  _Float16* x1    = Hh(1048576);
  _Float16* spk2  = Hh(2097152);
  _Float16* x2    = Hh(524288);
  _Float16* spk3  = Hh(1048576);
  _Float16* x3    = Hh(262144);      // flattened [B,2048]
  _Float16* spk4  = Hh(32768);
  _Float16* spk5  = Hh(16384);
  _Float16* w1h   = Hh(32 * 32);
  _Float16* w2h   = Hh(64 * 288);
  _Float16* w3h   = Hh(128 * 576);
  _Float16* wf1h  = Hh(256 * 2048);
  _Float16* wf2h  = Hh(128 * 256);
  _Float16* wf3h  = Hh(16 * 128);

  const long ZN = 4194304L + 2097152 + 1048576 + 32768 + 16384 + 1280 + 16;
  k_zero<<<(int)((ZN + 255) / 256), 256, 0, stream>>>(mem1, ZN);

  // weights -> padded f16
  k_cvt_pad<<<(32 * 32 + 255) / 256, 256, 0, stream>>>(w1, w1h, 32, 32, 27, 32);
  k_cvt_pad<<<(64 * 288 + 255) / 256, 256, 0, stream>>>(w2, w2h, 64, 64, 288, 288);
  k_cvt_pad<<<(128 * 576 + 255) / 256, 256, 0, stream>>>(w3, w3h, 128, 128, 576, 576);
  k_cvt_pad<<<(256 * 2048 + 255) / 256, 256, 0, stream>>>(wf1, wf1h, 256, 256, 2048, 2048);
  k_cvt_pad<<<(128 * 256 + 255) / 256, 256, 0, stream>>>(wf2, wf2h, 128, 128, 256, 256);
  k_cvt_pad<<<(16 * 128 + 255) / 256, 256, 0, stream>>>(wf3, wf3h, 10, 16, 128, 128);

  for (int t = 0; t < 25; ++t) {
    k_spikegen<<<393216 / 256, 256, 0, stream>>>(x, sin_t, t, 393216);

    // conv1: M=131072 (8192 tiles) x N=32 (2 tiles) -> 16384 waves
    k_conv_wmma<3, 32, 32, 32><<<16384 / 4, 128, 0, stream>>>(sin_t, w1h, b1, cur, 32);
    k_lif<<<4194304 / 256, 256, 0, stream>>>(cur, mem1, spk1, acc + 0, 4194304);
    k_pool<<<1048576 / 256, 256, 0, stream>>>(spk1, x1, acc + 6, 32, 16, 16, 1048576);

    // conv2: M=32768 (2048) x N=64 (4) -> 8192 waves
    k_conv_wmma<32, 16, 16, 288><<<8192 / 4, 128, 0, stream>>>(x1, w2h, b2, cur, 64);
    k_lif<<<2097152 / 256, 256, 0, stream>>>(cur, mem2, spk2, acc + 1, 2097152);
    k_pool<<<524288 / 256, 256, 0, stream>>>(spk2, x2, acc + 7, 64, 8, 8, 524288);

    // conv3: M=8192 (512) x N=128 (8) -> 4096 waves
    k_conv_wmma<64, 8, 8, 576><<<4096 / 4, 128, 0, stream>>>(x2, w3h, b3, cur, 128);
    k_lif<<<1048576 / 256, 256, 0, stream>>>(cur, mem3, spk3, acc + 2, 1048576);
    k_pool<<<262144 / 256, 256, 0, stream>>>(spk3, x3, acc + 8, 128, 4, 4, 262144);

    // fc1: M-tiles=8 blocks; N=256 (16 tiles, 4/wave); K=2048 in 4 TDM chunks
    k_fc_wmma<<<8, 128, 0, stream>>>(x3, wf1h, bf1, cur, 256, 2048, 512, 7, 256, 256);
    k_lif<<<32768 / 256, 256, 0, stream>>>(cur, mem4, spk4, acc + 3, 32768);

    // fc2: 8 blocks; N=128 (8 tiles, 2/wave); K=256 single chunk
    k_fc_wmma<<<8, 128, 0, stream>>>(spk4, wf2h, bf2, cur, 128, 256, 256, 6, 128, 128);
    k_lif<<<16384 / 256, 256, 0, stream>>>(cur, mem5, spk5, acc + 4, 16384);

    // fc3: 8 blocks; N=16 (1 tile, wave 0 only); K=128 single chunk
    k_fc_wmma<<<8, 128, 0, stream>>>(spk5, wf3h, bf3, cur, 16, 128, 128, 5, 10, 10);
    k_lif_out<<<(1280 + 255) / 256, 256, 0, stream>>>(cur, mem6,
                                                      out + (long)t * 1280,
                                                      out + 32000 + (long)t * 1280,
                                                      acc + 5, 1280);
  }

  k_finalize<<<1, 32, 0, stream>>>(acc, out + 64000);
}
